// IMUKinematicNetwork_53231824666776
// MI455X (gfx1250) — matrix operand order repaired and checked by reading, not compiled, not run on hardware
//
#include <hip/hip_runtime.h>

// ---------------------------------------------------------------------------
// IMU GRU network, fully fused for MI455X (gfx1250, wave32, WMMA bf16).
//
// Sizes (compile-time): B=4096, T=512, IN=17, HID=512, LAT=256, G=3*HID=1536.
// Strategy: 256 workgroups, each owns 16 rows of B and runs the whole T loop
// with h double-buffered in LDS (bf16). All GEMMs use
// v_wmma_f32_16x16x32_bf16; weights are pre-converted to bf16 in d_ws by a
// prep kernel and streamed from L2 as B-fragments (row-major [n][k], 32
// contiguous bytes per lane -> no transpose needed).
// ---------------------------------------------------------------------------

typedef __bf16 v8bf  __attribute__((ext_vector_type(8)));
typedef __bf16 v16bf __attribute__((ext_vector_type(16)));
typedef float  v8f   __attribute__((ext_vector_type(8)));

static constexpr int Bn   = 4096;
static constexpr int Tn   = 512;
static constexpr int IN   = 17;
static constexpr int HID  = 512;
static constexpr int LAT  = 256;
static constexpr int G    = 3 * HID;     // 1536 gate rows
static constexpr int KPAD = 288;         // w_ih K (257) padded to 9*32
static constexpr int MT   = 16;          // batch rows per workgroup

// LDS row strides (bf16 elems): multiple of 8 (16B-aligned rows for
// ds_load_b128) and (stride/2) % 64 != 0 so the 16 lanes of an A-frag
// (same k, consecutive m) land on distinct banks.
static constexpr int HSTR = 520;   // h rows      (520/2 % 64 = 4)
static constexpr int ZSTR = 264;   // z1 rows     (264/2 % 64 = 4)
static constexpr int GSTR = 296;   // gin rows    (296/2 % 64 = 20)

union ABu { v16bf v; v8bf h[2]; };

// A-frag: 16x32 bf16 tile from LDS. Lane L: M = L&15, kbase = 16*(L>>4);
// elems 0..15 = K = k0+kbase+{0..15}.
__device__ __forceinline__ v16bf load_afrag(const __bf16* base, int stride,
                                            int k0, int lane) {
  const int m  = lane & 15;
  const int ks = (lane >> 4) << 4;           // 0 or 16
  const __bf16* p = base + m * stride + k0 + ks;
  ABu u;
  u.h[0] = *(const v8bf*)(p);
  u.h[1] = *(const v8bf*)(p + 8);
  return u.v;
}

// B-frag: 32x16 bf16 tile from row-major [n][k] weights in global.
// Lane L: N = n0 + (L&15), K = k0 + 16*(L>>4) + {0..15} -> 32 contiguous B.
__device__ __forceinline__ v16bf load_bfrag(const __bf16* w, int stride,
                                            int n0, int k0, int lane) {
  const int n  = n0 + (lane & 15);
  const int kb = k0 + ((lane >> 4) << 4);    // +0 or +16
  return *(const v16bf*)(w + n * stride + kb);
}

__device__ __forceinline__ v8f wmma_bf16(v16bf a, v16bf b, v8f c) {
  return __builtin_amdgcn_wmma_f32_16x16x32_bf16(
      /*neg_a=*/false, a, /*neg_b=*/false, b,
      /*c_mod=*/(short)0, c, /*reuse_a=*/false, /*reuse_b=*/false);
}

__device__ __forceinline__ float sigf(float x) {
  return 1.0f / (1.0f + __expf(-x));
}

// ---------------------------------------------------------------------------
// Prep: convert weights f32 -> bf16 into workspace (w_ih zero-padded to KPAD)
// ---------------------------------------------------------------------------
__global__ void prep_weights(const float* __restrict__ w_hh,
                             const float* __restrict__ w_ih,
                             const float* __restrict__ enc_w2,
                             __bf16* __restrict__ whh,
                             __bf16* __restrict__ wih,
                             __bf16* __restrict__ ew2) {
  const int stride = gridDim.x * blockDim.x;
  const int i0 = blockIdx.x * blockDim.x + threadIdx.x;
  for (int j = i0; j < G * HID; j += stride) whh[j] = (__bf16)w_hh[j];
  for (int j = i0; j < G * KPAD; j += stride) {
    const int n = j / KPAD, k = j % KPAD;
    wih[j] = (k < LAT + 1) ? (__bf16)w_ih[n * (LAT + 1) + k] : (__bf16)0.0f;
  }
  for (int j = i0; j < LAT * LAT; j += stride) ew2[j] = (__bf16)enc_w2[j];
}

// ---------------------------------------------------------------------------
// Fused recurrent kernel: grid = B/MT = 256 blocks, 256 threads (8 waves)
// ---------------------------------------------------------------------------
__global__ __launch_bounds__(256)
void imu_fused(const float* __restrict__ x,   const float* __restrict__ h0,
               const float* __restrict__ w1,  const float* __restrict__ b1,
               const float* __restrict__ b2,
               const __bf16* __restrict__ ew2,
               const __bf16* __restrict__ wih,
               const __bf16* __restrict__ whh,
               const float* __restrict__ b_ih, const float* __restrict__ b_hh,
               const float* __restrict__ qw,  const float* __restrict__ qb,
               const float* __restrict__ vw,  const float* __restrict__ vb,
               const float* __restrict__ cw,  const float* __restrict__ cb,
               float* __restrict__ out) {
  __shared__ __align__(16) __bf16 hbuf[2][MT * HSTR];  // h state, double buf
  __shared__ __align__(16) __bf16 z1s[MT * ZSTR];      // encoder layer 1
  __shared__ __align__(16) __bf16 gins[MT * GSTR];     // [lat | dt | 0-pad]
  __shared__ float xt[MT][IN + 1];                     // x tile (f32)
  __shared__ float hscr[MT * 16 * 10];                 // head split-K partials

  const int tid  = threadIdx.x;
  const int lane = tid & 31;
  const int wv   = tid >> 5;                           // wave 0..7
  const int b0   = blockIdx.x * MT;

  // Init h state from h0 (bf16) and zero the gin K-pad region once.
  for (int i = tid; i < MT * HID; i += 256) {
    const int m = i / HID, k = i % HID;
    hbuf[0][m * HSTR + k] = (__bf16)h0[(b0 + m) * HID + k];
  }
  for (int i = tid; i < MT * (KPAD - (LAT + 1)); i += 256) {
    const int m = i / (KPAD - (LAT + 1));
    const int c = (LAT + 1) + i % (KPAD - (LAT + 1));
    gins[m * GSTR + c] = (__bf16)0.0f;
  }
  float dp0 = 0.0f, dp1 = 0.0f, dp2 = 0.0f;
  __syncthreads();

  for (int t = 0; t < Tn; ++t) {
    const int src = t & 1, dst = src ^ 1;

    // ---- stage x tile (only HBM traffic in the loop) --------------------
    for (int i = tid; i < MT * IN; i += 256) {
      const int m = i / IN, k = i % IN;
      xt[m][k] = x[((b0 + m) * Tn + t) * IN + k];
    }
    __syncthreads();

    // ---- encoder layer 1 (K=17, VALU): z1 = relu(x @ w1^T + b1) ---------
    {
      const int c = tid;  // 256 threads -> one latent column each
      for (int m = 0; m < MT; ++m) {
        float s = b1[c];
#pragma unroll
        for (int k = 0; k < IN; ++k) s += xt[m][k] * w1[c * IN + k];
        z1s[m * ZSTR + c] = (__bf16)fmaxf(s, 0.0f);
      }
      if (tid < MT) gins[tid * GSTR + LAT] = (__bf16)xt[tid][IN - 1];  // dt
    }
    __syncthreads();

    // ---- encoder layer 2 (WMMA, K=256): lat -> gins[:, 0:256] ----------
    for (int nt = wv * 2; nt < wv * 2 + 2; ++nt) {
      const int nc = nt * 16;
      v8f acc = {};
      for (int kt = 0; kt < LAT / 32; ++kt) {
        const v16bf a  = load_afrag(z1s, ZSTR, kt * 32, lane);
        const v16bf bm = load_bfrag(ew2, LAT, nc, kt * 32, lane);
        acc = wmma_bf16(a, bm, acc);
      }
      const int col = nc + (lane & 15);
      const float bb = b2[col];
#pragma unroll
      for (int r = 0; r < 8; ++r) {
        const int m = r + ((lane >> 4) << 3);
        gins[m * GSTR + col] = (__bf16)fmaxf(acc[r] + bb, 0.0f);
      }
    }
    __syncthreads();

    // ---- GRU gates (WMMA): r/z fuse gi+gh; i_n and h_n kept separate ----
    for (int nt = wv * 4; nt < wv * 4 + 4; ++nt) {
      const int nc = nt * 16;
      v8f ar = {}, az = {}, ain = {}, ahn = {};
      // gi part: gin(K=288) x w_ih
      for (int kt = 0; kt < KPAD / 32; ++kt) {
        const v16bf a  = load_afrag(gins, GSTR, kt * 32, lane);
        const v16bf br = load_bfrag(wih, KPAD, 0 * HID + nc, kt * 32, lane);
        const v16bf bz = load_bfrag(wih, KPAD, 1 * HID + nc, kt * 32, lane);
        const v16bf bn = load_bfrag(wih, KPAD, 2 * HID + nc, kt * 32, lane);
        ar  = wmma_bf16(a, br, ar);
        az  = wmma_bf16(a, bz, az);
        ain = wmma_bf16(a, bn, ain);
      }
      // gh part: h(K=512) x w_hh
      for (int kt = 0; kt < HID / 32; ++kt) {
        const v16bf a  = load_afrag(hbuf[src], HSTR, kt * 32, lane);
        const v16bf br = load_bfrag(whh, HID, 0 * HID + nc, kt * 32, lane);
        const v16bf bz = load_bfrag(whh, HID, 1 * HID + nc, kt * 32, lane);
        const v16bf bn = load_bfrag(whh, HID, 2 * HID + nc, kt * 32, lane);
        ar  = wmma_bf16(a, br, ar);
        az  = wmma_bf16(a, bz, az);
        ahn = wmma_bf16(a, bn, ahn);
      }
      const int col = nc + (lane & 15);
      const float bir = b_ih[col], biz = b_ih[HID + col], bin_ = b_ih[2 * HID + col];
      const float bhr = b_hh[col], bhz = b_hh[HID + col], bhn  = b_hh[2 * HID + col];
#pragma unroll
      for (int r = 0; r < 8; ++r) {
        const int m  = r + ((lane >> 4) << 3);
        const float rg = sigf(ar[r] + bir + bhr);
        const float zg = sigf(az[r] + biz + bhz);
        const float nn = tanhf(ain[r] + bin_ + rg * (ahn[r] + bhn));
        const float ho = (float)hbuf[src][m * HSTR + col];
        hbuf[dst][m * HSTR + col] = (__bf16)((1.0f - zg) * nn + zg * ho);
      }
    }
    __syncthreads();

    // ---- heads (VALU, split-K=16): 10 dots of 512 per row --------------
    {
      const int m = tid >> 4, p = tid & 15, kb = p * 32;
      float s[10];
#pragma unroll
      for (int j = 0; j < 10; ++j) s[j] = 0.0f;
      for (int k = 0; k < 32; ++k) {
        const int kk = kb + k;
        const float hk = (float)hbuf[dst][m * HSTR + kk];
        s[0] += hk * qw[0 * HID + kk]; s[1] += hk * qw[1 * HID + kk];
        s[2] += hk * qw[2 * HID + kk]; s[3] += hk * qw[3 * HID + kk];
        s[4] += hk * vw[0 * HID + kk]; s[5] += hk * vw[1 * HID + kk];
        s[6] += hk * vw[2 * HID + kk];
        s[7] += hk * cw[0 * HID + kk]; s[8] += hk * cw[1 * HID + kk];
        s[9] += hk * cw[2 * HID + kk];
      }
#pragma unroll
      for (int j = 0; j < 10; ++j) hscr[(m * 16 + p) * 10 + j] = s[j];
    }
    __syncthreads();

    if ((tid & 15) == 0) {
      const int m = tid >> 4;
      float hv[10];
#pragma unroll
      for (int j = 0; j < 10; ++j) {
        float s = 0.0f;
        for (int p = 0; p < 16; ++p) s += hscr[(m * 16 + p) * 10 + j];
        hv[j] = s;
      }
      float q0 = hv[0] + qb[0], q1 = hv[1] + qb[1];
      float q2 = hv[2] + qb[2], q3 = hv[3] + qb[3];
      float nrm = fmaxf(sqrtf(q0 * q0 + q1 * q1 + q2 * q2 + q3 * q3), 1e-12f);
      const float w = q0 / nrm, xq = q1 / nrm, yq = q2 / nrm, zq = q3 / nrm;
      const float vx = hv[4] + vb[0], vy = hv[5] + vb[1], vz = hv[6] + vb[2];
      const float ex = hv[7] + cb[0], ey = hv[8] + cb[1], ez = hv[9] + cb[2];
      const float axl = xt[m][0], ayl = xt[m][1], azl = xt[m][2];
      const float dt = xt[m][IN - 1];
      const float xx = xq * xq, yy = yq * yq, zz = zq * zq;
      const float xy = xq * yq, xz = xq * zq, yz = yq * zq;
      const float wx = w * xq, wy = w * yq, wz = w * zq;
      const float aw0 = (1 - 2 * (yy + zz)) * axl + 2 * (xy - wz) * ayl + 2 * (xz + wy) * azl;
      const float aw1 = 2 * (xy + wz) * axl + (1 - 2 * (xx + zz)) * ayl + 2 * (yz - wx) * azl;
      const float aw2 = 2 * (xz - wy) * axl + 2 * (yz + wx) * ayl + (1 - 2 * (xx + yy)) * azl;
      const float hdt = 0.5f * dt * dt;
      dp0 += vx * dt + aw0 * hdt + ex;
      dp1 += vy * dt + aw1 * hdt + ey;
      dp2 += vz * dt + aw2 * hdt + ez;
    }
    __syncthreads();  // protect xt / hscr reuse next iteration
  }

  if ((tid & 15) == 0) {
    const int m = tid >> 4;
    out[(b0 + m) * 3 + 0] = dp0;
    out[(b0 + m) * 3 + 1] = dp1;
    out[(b0 + m) * 3 + 2] = dp2;
  }
}

// ---------------------------------------------------------------------------
extern "C" void kernel_launch(void* const* d_in, const int* in_sizes, int n_in,
                              void* d_out, int out_size, void* d_ws, size_t ws_size,
                              hipStream_t stream) {
  const float* x      = (const float*)d_in[0];
  const float* h0     = (const float*)d_in[1];
  const float* enc_w1 = (const float*)d_in[2];
  const float* enc_b1 = (const float*)d_in[3];
  const float* enc_w2 = (const float*)d_in[4];
  const float* enc_b2 = (const float*)d_in[5];
  const float* w_ih   = (const float*)d_in[6];
  const float* w_hh   = (const float*)d_in[7];
  const float* b_ih   = (const float*)d_in[8];
  const float* b_hh   = (const float*)d_in[9];
  const float* q_w    = (const float*)d_in[10];
  const float* q_b    = (const float*)d_in[11];
  const float* v_w    = (const float*)d_in[12];
  const float* v_b    = (const float*)d_in[13];
  const float* c_w    = (const float*)d_in[14];
  const float* c_b    = (const float*)d_in[15];

  // Workspace layout (bf16 weights, ~2.6 MB, all 32B-aligned):
  __bf16* whh = (__bf16*)d_ws;                 // [G][HID]
  __bf16* wih = whh + (size_t)G * HID;         // [G][KPAD] (zero-padded K)
  __bf16* ew2 = wih + (size_t)G * KPAD;        // [LAT][LAT]

  prep_weights<<<1024, 256, 0, stream>>>(w_hh, w_ih, enc_w2, whh, wih, ew2);

  imu_fused<<<Bn / MT, 256, 0, stream>>>(
      x, h0, enc_w1, enc_b1, enc_b2, ew2, wih, whh, b_ih, b_hh,
      q_w, q_b, v_w, v_b, c_w, c_b, (float*)d_out);
}
